// AthenaSA_42167988912761
// MI455X (gfx1250) — matrix operand order repaired and checked by reading, compiled, not run on hardware
//
#include <hip/hip_runtime.h>
#include <cstdint>
#include <cstddef>

// Problem constants (match reference)
#define EMB   2048
#define KEYD  128
#define HEADD 128
#define NH    16
#define BB    2
#define SS    2048
#define MROWS (BB * SS)                      // 4096
#define QKVN  (2 * KEYD * NH + HEADD * NH)   // 6144

typedef __bf16 v16bf __attribute__((ext_vector_type(16)));
typedef __bf16 v8bf  __attribute__((ext_vector_type(8)));
typedef float  v8f   __attribute__((ext_vector_type(8)));

union BF16x16 { v16bf v; v8bf h[2]; };

__device__ __forceinline__ unsigned short f2bf(float f) {
  unsigned int u = __float_as_uint(f);
  u += 0x7FFFu + ((u >> 16) & 1u);           // round-to-nearest-even
  return (unsigned short)(u >> 16);
}
__device__ __forceinline__ float bf2f(unsigned short h) {
  return __uint_as_float(((unsigned int)h) << 16);
}

// CDNA5 async global->LDS copy (ASYNCcnt-tracked, no VGPR round trip).
__device__ __forceinline__ void async_ld16(unsigned lds_off, const void* gptr) {
  asm volatile("global_load_async_to_lds_b128 %0, %1, off"
               :: "v"(lds_off), "v"(gptr) : "memory");
}
__device__ __forceinline__ void wait_async0() {
  asm volatile("s_wait_asynccnt 0x0" ::: "memory");
}
__device__ __forceinline__ unsigned lds_off32(const void* p) {
  return (unsigned)(size_t)p;                // LDS flat addr: low 32 bits = offset
}

// ---------------------------------------------------------------------------
// Kernel 1: RMSNorm over EMB=2048, output bf16.  One block per row.
// ---------------------------------------------------------------------------
__global__ __launch_bounds__(256) void k_rmsnorm_bf16(
    const float* __restrict__ x, unsigned short* __restrict__ y) {
  __shared__ float red[8];
  const int row = blockIdx.x;
  const int t   = threadIdx.x;
  const float* xr = x + (size_t)row * EMB;
  float4 a = ((const float4*)xr)[t];
  float4 b = ((const float4*)xr)[t + 256];
  float ss = a.x * a.x + a.y * a.y + a.z * a.z + a.w * a.w +
             b.x * b.x + b.y * b.y + b.z * b.z + b.w * b.w;
#pragma unroll
  for (int off = 16; off > 0; off >>= 1) ss += __shfl_xor(ss, off, 32);
  if ((t & 31) == 0) red[t >> 5] = ss;
  __syncthreads();
  float tot = 0.f;
#pragma unroll
  for (int i = 0; i < 8; ++i) tot += red[i];
  const float r = rsqrtf(tot / (float)EMB + 1e-5f);
  unsigned short* yr = y + (size_t)row * EMB;
  const int i0 = t * 4, i1 = (t + 256) * 4;
  yr[i0 + 0] = f2bf(a.x * r); yr[i0 + 1] = f2bf(a.y * r);
  yr[i0 + 2] = f2bf(a.z * r); yr[i0 + 3] = f2bf(a.w * r);
  yr[i1 + 0] = f2bf(b.x * r); yr[i1 + 1] = f2bf(b.y * r);
  yr[i1 + 2] = f2bf(b.z * r); yr[i1 + 3] = f2bf(b.w * r);
}

// ---------------------------------------------------------------------------
// Kernel 2: generic f32 -> bf16 conversion (for the weight matrices).
// ---------------------------------------------------------------------------
__global__ __launch_bounds__(256) void k_f32_to_bf16(
    const float* __restrict__ x, unsigned short* __restrict__ y, int n) {
  int i = blockIdx.x * blockDim.x + threadIdx.x;
  const int stride = gridDim.x * blockDim.x;
  for (; i < n; i += stride) y[i] = f2bf(x[i]);
}

// ---------------------------------------------------------------------------
// Kernel 3: bf16 WMMA GEMM:  C[M,N] = A[M,K] * Bw[N,K]^T
//   128x128 tile, BK=64, double-buffered LDS via GLOBAL_LOAD_ASYNC_TO_LDS_B128.
//   Fragment loads hoisted -> 12 ds_load burst, one wait, 8-WMMA burst.
//   Epilogue stages C through the (now dead) LDS tile for coalesced stores.
//   mode 0: bf16 C.   mode 1: f32 C = resid + acc.
// ---------------------------------------------------------------------------
#define TSZ (128 * 72)   // one staging buffer in halves (64 + 8 pad per row)

__global__ __launch_bounds__(256) void k_gemm_bf16_wmma(
    const unsigned short* __restrict__ A,    // [M,K] bf16 row-major
    const unsigned short* __restrict__ Bw,   // [N,K] bf16 row-major
    unsigned short* __restrict__ Cbf,        // mode 0 output
    float* __restrict__ Cf,                  // mode 1 output
    const float* __restrict__ resid,         // mode 1 residual
    int M, int N, int K, int mode) {
  // Layout: [0,TSZ) = A buf0, [TSZ,2T) = A buf1, [2T,3T) = B buf0, [3T,4T) = B buf1
  __shared__ __align__(16) unsigned short smem[4 * TSZ];   // 73728 B

  const int tid  = threadIdx.x;
  const int w    = tid >> 5;
  const int lane = tid & 31;
  const int hi   = lane >> 4;                // half-wave index (ISA frag layouts)
  const int lr   = lane & 15;
  const int wm   = w >> 2, wn = w & 3;
  const int m0   = blockIdx.y * 128, n0 = blockIdx.x * 128;
  (void)M;

  // Per-thread staging slots: 4 x 16B chunks per matrix per tile.
  int srow[4], sch[4];
  unsigned ldsA[2][4], ldsB[2][4];
#pragma unroll
  for (int it = 0; it < 4; ++it) {
    const int i = it * 256 + tid;            // 0..1023
    srow[it] = i >> 3;                       // 0..127
    sch[it]  = (i & 7) * 8;                  // half offset 0..56
#pragma unroll
    for (int bb = 0; bb < 2; ++bb) {
      ldsA[bb][it] = lds_off32(&smem[bb * TSZ + srow[it] * 72 + sch[it]]);
      ldsB[bb][it] = lds_off32(&smem[(2 + bb) * TSZ + srow[it] * 72 + sch[it]]);
    }
  }

  v8f acc[4][2];
  {
    v8f zf = {};
#pragma unroll
    for (int mt = 0; mt < 4; ++mt)
#pragma unroll
      for (int nt = 0; nt < 2; ++nt) acc[mt][nt] = zf;
  }

  const int ntiles = K >> 6;

  auto issue = [&](int t, int bb) {
    const int k0 = t * 64;
#pragma unroll
    for (int it = 0; it < 4; ++it) {
      async_ld16(ldsA[bb][it], &A[(size_t)(m0 + srow[it]) * K + k0 + sch[it]]);
      async_ld16(ldsB[bb][it], &Bw[(size_t)(n0 + srow[it]) * K + k0 + sch[it]]);
    }
  };

  issue(0, 0);
  for (int t = 0; t < ntiles; ++t) {
    wait_async0();                 // own tile-t transfers complete
    __syncthreads();               // all waves' tile-t data visible; buf (t+1)&1 free
    if (t + 1 < ntiles) issue(t + 1, (t + 1) & 1);   // overlap with compute

    const unsigned short* as = smem + (t & 1) * TSZ;
    const unsigned short* bs = smem + (2 + (t & 1)) * TSZ;
#pragma unroll
    for (int ks = 0; ks < 2; ++ks) {         // two WMMA K-steps per tile
      BF16x16 bfrag[2], afrag[4];
      // B fragments (32x16): lane lr = column n, 16 contiguous K at (hi*16)
#pragma unroll
      for (int nt = 0; nt < 2; ++nt) {
        const unsigned short* bp = &bs[(wn * 32 + nt * 16 + lr) * 72 + ks * 32 + hi * 16];
        bfrag[nt].h[0] = *(const v8bf*)bp;
        bfrag[nt].h[1] = *(const v8bf*)(bp + 8);
      }
      // A fragments (16x32): lane lr = row m, 16 contiguous K at (hi*16)
#pragma unroll
      for (int mt = 0; mt < 4; ++mt) {
        const unsigned short* ap = &as[(wm * 64 + mt * 16 + lr) * 72 + ks * 32];
        afrag[mt].h[0] = *(const v8bf*)(ap + hi * 16);
        afrag[mt].h[1] = *(const v8bf*)(ap + hi * 16 + 8);
      }
      // 8 back-to-back WMMAs
#pragma unroll
      for (int mt = 0; mt < 4; ++mt)
#pragma unroll
        for (int nt = 0; nt < 2; ++nt)
          acc[mt][nt] = __builtin_amdgcn_wmma_f32_16x16x32_bf16(
              false, afrag[mt].v, false, bfrag[nt].v, (short)0, acc[mt][nt], false, false);
    }
  }

  // Coalesced epilogue through LDS (staging buffers are dead now).
  __syncthreads();
  if (mode == 0) {
    unsigned short* ct = smem;               // [128][136] halves, 34816 B
#pragma unroll
    for (int mt = 0; mt < 4; ++mt)
#pragma unroll
      for (int nt = 0; nt < 2; ++nt)
#pragma unroll
        for (int v = 0; v < 8; ++v)
          ct[(wm * 64 + mt * 16 + v + 8 * hi) * 136 + wn * 32 + nt * 16 + lr] =
              f2bf(acc[mt][nt][v]);
    __syncthreads();
#pragma unroll
    for (int it = 0; it < 8; ++it) {
      const int i = it * 256 + tid;          // 0..2047 chunks of 8 halves
      const int row = i >> 4, c = (i & 15) * 8;
      *(uint4*)&Cbf[(size_t)(m0 + row) * N + n0 + c] = *(const uint4*)&ct[row * 136 + c];
    }
  } else {
    float* ct = (float*)smem;                // [128][132] floats, 67584 B
#pragma unroll
    for (int mt = 0; mt < 4; ++mt)
#pragma unroll
      for (int nt = 0; nt < 2; ++nt)
#pragma unroll
        for (int v = 0; v < 8; ++v)
          ct[(wm * 64 + mt * 16 + v + 8 * hi) * 132 + wn * 32 + nt * 16 + lr] =
              acc[mt][nt][v];
    __syncthreads();
#pragma unroll
    for (int it = 0; it < 16; ++it) {
      const int i = it * 256 + tid;          // 0..4095 chunks of 4 floats
      const int row = i >> 5, c = (i & 31) * 4;
      const size_t g = (size_t)(m0 + row) * N + n0 + c;
      float4 rv = *(const float4*)&resid[g];
      float4 cv = *(const float4*)&ct[row * 132 + c];
      cv.x += rv.x; cv.y += rv.y; cv.z += rv.z; cv.w += rv.w;
      *(float4*)&Cf[g] = cv;
    }
  }
}

// ---------------------------------------------------------------------------
// Kernel 4: RoPE + rearrange:
//   qkv [4096, 6144] bf16 -> Q,K [B,H,S,128] bf16 (rotated), Vt [B,H,128,S] bf16
// ---------------------------------------------------------------------------
__global__ __launch_bounds__(256) void k_rope_rearrange(
    const unsigned short* __restrict__ qkv,
    const float* __restrict__ cosb, const float* __restrict__ sinb,
    unsigned short* __restrict__ Qo, unsigned short* __restrict__ Ko,
    unsigned short* __restrict__ Vto) {
  const int row = blockIdx.x;
  const int b = row >> 11, s = row & (SS - 1);
  const unsigned short* qr = qkv + (size_t)row * QKVN;
  const float* c0 = cosb + (size_t)s * KEYD;                       // cos_buffer[0]
  const float* s0 = sinb + (size_t)s * KEYD;
  const float* c1 = cosb + (size_t)SS * KEYD + (size_t)s * KEYD;   // cos_buffer[1]
  const float* s1 = sinb + (size_t)SS * KEYD + (size_t)s * KEYD;
  for (int i = threadIdx.x; i < NH * KEYD; i += 256) {
    const int h = i >> 7, d = i & 127, dp = (d + 64) & 127;
    const float cq = c0[d], sq = s0[d], ck = c1[d], sk = s1[d];
    const float xq  = bf2f(qr[i]);
    const float xqp = bf2f(qr[(i - d) + dp]);                       // swapped half
    const float xk  = bf2f(qr[2048 + i]);
    const float xkp = bf2f(qr[2048 + (i - d) + dp]);
    const float xv  = bf2f(qr[4096 + i]);
    const size_t bh = (size_t)(b * NH + h);
    Qo[(bh * SS + s) * KEYD + d]   = f2bf(cq * xq + sq * xqp);
    Ko[(bh * SS + s) * KEYD + d]   = f2bf(ck * xk + sk * xkp);
    Vto[(bh * HEADD + d) * SS + s] = f2bf(xv);
  }
}

// ---------------------------------------------------------------------------
// Kernel 5: causal flash attention, one block per (q-tile of 64, batch*head).
//   4 waves; each wave owns 16 q rows -> fully wave-local online softmax.
//   QK^T and P*V both via v_wmma_f32_16x16x32_bf16 (f32 accumulate).
// ---------------------------------------------------------------------------
__global__ __launch_bounds__(128) void k_attn_wmma(
    const unsigned short* __restrict__ Q,
    const unsigned short* __restrict__ Kx,
    const unsigned short* __restrict__ Vt,
    unsigned short* __restrict__ O) {
  __shared__ unsigned short pls[4][16][40];   // per-wave P tile (16x32, padded)
  const int bh = blockIdx.y;
  const int b = bh >> 4, h = bh & 15;
  const int tid  = threadIdx.x;
  const int w    = tid >> 5;
  const int lane = tid & 31;
  const int hi   = lane >> 4;
  const int lr   = lane & 15;
  const int q0   = blockIdx.x * 64 + w * 16;
  const size_t base = (size_t)bh * SS * 128;
  const unsigned short* Qb = Q + base;
  const unsigned short* Kb = Kx + base;
  const unsigned short* Vb = Vt + base;

  // Q fragments: 4 d-chunks of 32, A-layout
  BF16x16 qf[4];
  {
    const unsigned short* qr = Qb + (size_t)(q0 + lr) * 128;
#pragma unroll
    for (int dc = 0; dc < 4; ++dc) {
      qf[dc].h[0] = *(const v8bf*)(qr + dc * 32 + hi * 16);
      qf[dc].h[1] = *(const v8bf*)(qr + dc * 32 + hi * 16 + 8);
    }
  }

  v8f Of[8];
  {
    v8f zf = {};
#pragma unroll
    for (int d = 0; d < 8; ++d) Of[d] = zf;
  }
  float mrun[8], lsum[8];
#pragma unroll
  for (int v = 0; v < 8; ++v) { mrun[v] = -3.0e38f; lsum[v] = 0.f; }

  const float scale = 0.08838834764831845f;   // 1/sqrt(128)
  const int nkb = (q0 + 16 + 31) >> 5;        // causal: keys <= q0+15

  for (int kb = 0; kb < nkb; ++kb) {
    const int kbase = kb * 32;

    // Load all 8 K fragments, then run the two 4-deep WMMA chains.
    BF16x16 kf0[4], kf1[4];
#pragma unroll
    for (int dc = 0; dc < 4; ++dc) {
      const unsigned short* kp0 = Kb + (size_t)(kbase + lr) * 128 + dc * 32 + hi * 16;
      const unsigned short* kp1 = Kb + (size_t)(kbase + 16 + lr) * 128 + dc * 32 + hi * 16;
      kf0[dc].h[0] = *(const v8bf*)kp0; kf0[dc].h[1] = *(const v8bf*)(kp0 + 8);
      kf1[dc].h[0] = *(const v8bf*)kp1; kf1[dc].h[1] = *(const v8bf*)(kp1 + 8);
    }
    v8f s0 = {}, s1 = {};
#pragma unroll
    for (int dc = 0; dc < 4; ++dc)
      s0 = __builtin_amdgcn_wmma_f32_16x16x32_bf16(false, qf[dc].v, false, kf0[dc].v,
                                                   (short)0, s0, false, false);
#pragma unroll
    for (int dc = 0; dc < 4; ++dc)
      s1 = __builtin_amdgcn_wmma_f32_16x16x32_bf16(false, qf[dc].v, false, kf1[dc].v,
                                                   (short)0, s1, false, false);

    // Online softmax update (row v+8*hi lives entirely in this 16-lane half)
#pragma unroll
    for (int v = 0; v < 8; ++v) {
      const int row = q0 + v + 8 * hi;
      float a  = s0[v] * scale; if (kbase + lr > row)      a  = -1e30f;
      float bb = s1[v] * scale; if (kbase + 16 + lr > row) bb = -1e30f;
      float mx = fmaxf(a, bb);
#pragma unroll
      for (int off = 1; off < 16; off <<= 1) mx = fmaxf(mx, __shfl_xor(mx, off, 32));
      const float mnew  = fmaxf(mrun[v], mx);
      const float alpha = __expf(mrun[v] - mnew);
      const float p0 = __expf(a - mnew), p1 = __expf(bb - mnew);
      float rs = p0 + p1;
#pragma unroll
      for (int off = 1; off < 16; off <<= 1) rs += __shfl_xor(rs, off, 32);
      lsum[v] = lsum[v] * alpha + rs;
      mrun[v] = mnew;
      pls[w][v + 8 * hi][lr]      = f2bf(p0);
      pls[w][v + 8 * hi][lr + 16] = f2bf(p1);
#pragma unroll
      for (int d = 0; d < 8; ++d) Of[d][v] *= alpha;
    }

    // Re-read P as an A-layout fragment (wave-private LDS; DS in-order per wave)
    BF16x16 pf;
    {
      const unsigned short* pp = &pls[w][lr][0];
      pf.h[0] = *(const v8bf*)(pp + hi * 16);
      pf.h[1] = *(const v8bf*)(pp + hi * 16 + 8);
    }

    // Load all 8 V fragments, then 8 independent accumulating WMMAs.
    BF16x16 vf[8];
#pragma unroll
    for (int dch = 0; dch < 8; ++dch) {
      const unsigned short* vp = Vb + (size_t)(dch * 16 + lr) * SS + kbase + hi * 16;
      vf[dch].h[0] = *(const v8bf*)vp; vf[dch].h[1] = *(const v8bf*)(vp + 8);
    }
#pragma unroll
    for (int dch = 0; dch < 8; ++dch)
      Of[dch] = __builtin_amdgcn_wmma_f32_16x16x32_bf16(false, pf.v, false, vf[dch].v,
                                                        (short)0, Of[dch], false, false);
  }

  // Normalize and store into [B, S, H*128] bf16 (GEMM-ready activation layout)
#pragma unroll
  for (int dch = 0; dch < 8; ++dch)
#pragma unroll
    for (int v = 0; v < 8; ++v) {
      const int srow = q0 + v + 8 * hi;
      const float o = Of[dch][v] / lsum[v];
      O[(size_t)(b * SS + srow) * (NH * HEADD) + h * HEADD + dch * 16 + lr] = f2bf(o);
    }
}

// ---------------------------------------------------------------------------
// Host-side launch
// d_in order: embeddings, cos_buffer, sin_buffer, causal_buffer, w_qkv, w_out
// ---------------------------------------------------------------------------
extern "C" void kernel_launch(void* const* d_in, const int* in_sizes, int n_in,
                              void* d_out, int out_size, void* d_ws, size_t ws_size,
                              hipStream_t stream) {
  (void)in_sizes; (void)n_in; (void)out_size; (void)ws_size;
  const float* emb  = (const float*)d_in[0];
  const float* cosb = (const float*)d_in[1];
  const float* sinb = (const float*)d_in[2];
  const float* wqkv = (const float*)d_in[4];
  const float* wout = (const float*)d_in[5];
  float* out = (float*)d_out;

  char* ws = (char*)d_ws;
  size_t off = 0;
  unsigned short* wqkv_bf = (unsigned short*)(ws + off); off += (size_t)QKVN * EMB * 2;   // 24 MB
  unsigned short* wout_bf = (unsigned short*)(ws + off); off += (size_t)EMB * EMB * 2;    //  8 MB
  unsigned short* norm_bf = (unsigned short*)(ws + off); off += (size_t)MROWS * EMB * 2;  // 16 MB
  unsigned short* qkv_bf  = (unsigned short*)(ws + off); off += (size_t)MROWS * QKVN * 2; // 48 MB
  unsigned short* Qb      = (unsigned short*)(ws + off); off += (size_t)BB * NH * SS * KEYD * 2;
  unsigned short* Kb      = (unsigned short*)(ws + off); off += (size_t)BB * NH * SS * KEYD * 2;
  unsigned short* Vtb     = (unsigned short*)(ws + off); off += (size_t)BB * NH * HEADD * SS * 2;
  unsigned short* attn_bf = qkv_bf;   // qkv dead after RoPE/rearrange -> reuse

  k_rmsnorm_bf16<<<MROWS, 256, 0, stream>>>(emb, norm_bf);
  k_f32_to_bf16<<<4096, 256, 0, stream>>>(wqkv, wqkv_bf, QKVN * EMB);
  k_f32_to_bf16<<<2048, 256, 0, stream>>>(wout, wout_bf, EMB * EMB);

  // qkv = normed @ w_qkv^T   (M=4096, N=6144, K=2048), bf16 out
  k_gemm_bf16_wmma<<<dim3(QKVN / 128, MROWS / 128), 256, 0, stream>>>(
      norm_bf, wqkv_bf, qkv_bf, nullptr, nullptr, MROWS, QKVN, EMB, 0);

  k_rope_rearrange<<<MROWS, 256, 0, stream>>>(qkv_bf, cosb, sinb, Qb, Kb, Vtb);

  k_attn_wmma<<<dim3(SS / 64, BB * NH), 128, 0, stream>>>(Qb, Kb, Vtb, attn_bf);

  // out = embeddings + attn @ w_out^T   (M=4096, N=2048, K=2048), f32 out
  k_gemm_bf16_wmma<<<dim3(EMB / 128, MROWS / 128), 256, 0, stream>>>(
      attn_bf, wout_bf, nullptr, out, emb, MROWS, EMB, EMB, 1);
}
